// RNN_11579231830577
// MI455X (gfx1250) — matrix operand order repaired and checked by reading, not compiled, not run on hardware
//
#include <hip/hip_runtime.h>
#include <hip/hip_bf16.h>

// Problem sizes (fixed by the reference)
#define BATCH 4
#define SEQ   4096
#define DDIM  2048
#define HDIM  4096
#define NCHUNK 64
#define CSIZE  64          // SEQ / NCHUNK

// LDS strides (pad to dodge bank conflicts; keep 16B alignment)
#define AST (DDIM + 8)     // A-tile row stride in bf16 elements (4112 B = 1028 dwords)
#define HST (HDIM + 8)     // H-tile row stride in bf16 elements

#define USE_TDM       1    // Tensor Data Mover for A-tile staging
#define USE_ASYNC_LDS 1    // fallback: GLOBAL_LOAD_ASYNC_TO_LDS_B128

typedef __attribute__((ext_vector_type(16))) __bf16 v16bf;
typedef __attribute__((ext_vector_type(8)))  float  v8f;
typedef __attribute__((ext_vector_type(4)))  unsigned int v4u;
typedef __attribute__((ext_vector_type(8)))  unsigned int v8u;

union AV { v16bf v; v4u q[2]; };
union B8 { v4u q; __bf16 e[8]; };

// Native converts: gfx1250 emits v_cvt_pk_bf16_f32 / fpext for these.
__device__ __forceinline__ __bf16 f2bf(float f)  { return (__bf16)f; }
__device__ __forceinline__ float  bf2f(__bf16 b) { return (float)b; }

// ---------------------------------------------------------------------------
// Kernel 1: transpose f32 [R,C] -> bf16 [C,R] (weights; coalesced via LDS tile)
// ---------------------------------------------------------------------------
__global__ void transpose_f32_to_bf16(const float* __restrict__ src,
                                      __bf16* __restrict__ dst,
                                      int R, int C) {
  __shared__ float tile[32][33];
  int c0 = blockIdx.x * 32, r0 = blockIdx.y * 32;
  int tx = threadIdx.x, ty = threadIdx.y;          // 32 x 8
  for (int i = ty; i < 32; i += 8) {
    int r = r0 + i, c = c0 + tx;
    tile[i][tx] = src[(size_t)r * C + c];
  }
  __syncthreads();
  for (int i = ty; i < 32; i += 8) {
    int r = c0 + i, c = r0 + tx;                   // dst is [C][R]
    dst[(size_t)r * R + c] = f2bf(tile[tx][i]);
  }
}

// ---------------------------------------------------------------------------
// Kernel 2: per-chunk column sums of X (pass 1 of blocked cumsum over S)
// ---------------------------------------------------------------------------
__global__ void chunk_sums(const float* __restrict__ X, float* __restrict__ sums) {
  int bc = blockIdx.x;                 // BATCH*NCHUNK blocks
  int b = bc / NCHUNK, c = bc % NCHUNK;
  const float* base = X + ((size_t)b * SEQ + (size_t)c * CSIZE) * DDIM;
  for (int d = threadIdx.x; d < DDIM; d += 256) {
    float acc = 0.f;
    for (int s = 0; s < CSIZE; ++s) acc += base[(size_t)s * DDIM + d];
    sums[((size_t)b * NCHUNK + c) * DDIM + d] = acc;
  }
}

// ---------------------------------------------------------------------------
// Kernel 3: prefix of chunk sums + in-chunk scan; emit Xc = cumsum_s(X) as bf16
// (valid because cumsum commutes with the linear map W1)
// ---------------------------------------------------------------------------
__global__ void scan_apply(const float* __restrict__ X,
                           const float* __restrict__ sums,
                           __bf16* __restrict__ Xc) {
  int bc = blockIdx.x;
  int b = bc / NCHUNK, c = bc % NCHUNK;
  const float* base = X + ((size_t)b * SEQ + (size_t)c * CSIZE) * DDIM;
  __bf16* obase     = Xc + ((size_t)b * SEQ + (size_t)c * CSIZE) * DDIM;
  for (int d = threadIdx.x; d < DDIM; d += 256) {
    float off = 0.f;
    for (int cc = 0; cc < c; ++cc) off += sums[((size_t)b * NCHUNK + cc) * DDIM + d];
    float acc = off;
    for (int s = 0; s < CSIZE; ++s) {
      acc += base[(size_t)s * DDIM + d];
      obase[(size_t)s * DDIM + d] = f2bf(acc);
    }
  }
}

// ---------------------------------------------------------------------------
// Kernel 4: fused  H = Xc@W1  ->  L2-normalize rows  ->  out = Hn@W2
// One 256-thread block (8 wave32) owns a 16-row slab. WMMA bf16, f32 accum,
// 4 N-tiles register-blocked per wave (4 independent WMMA chains).
// A-tile staged via TDM (TENSOR_LOAD_TO_LDS, iterate mode) -- one wave-level
// DMA instruction instead of 512 per-lane VMEM ops.
// ---------------------------------------------------------------------------
__launch_bounds__(256, 1)
__global__ void fused_mlp(const __bf16* __restrict__ Xc,
                          const __bf16* __restrict__ W1T,   // [HDIM][DDIM] bf16
                          const __bf16* __restrict__ W2T,   // [DDIM][HDIM] bf16
                          float* __restrict__ out) {
  extern __shared__ char smem[];
  __bf16* Abuf = (__bf16*)smem;                                   // 16*AST bf16
  __bf16* Hbuf = (__bf16*)(smem + (size_t)16 * AST * 2);          // 16*HST bf16
  float*  red  = (float*)(smem + (size_t)16 * AST * 2 + (size_t)16 * HST * 2);
  float*  rnrm = red + 256;

  const int tid  = threadIdx.x;
  const int wave = tid >> 5;          // gfx1250 is wave32
  const int lane = tid & 31;
  const int hh   = lane >> 4;         // half-wave select
  const int lm   = lane & 15;
  const size_t row0 = (size_t)blockIdx.x * 16;   // global row in [BATCH*SEQ)

  // ---- stage A tile (16 x DDIM bf16) into LDS ----
#if USE_TDM
  if (wave == 0) {
    // Tensor DMA descriptor (ISA ch.8): 1-row tile (1024 dwords), iterate 16x,
    // global stride = 1 row (1024 dwords), LDS stride = AST row (1028 dwords).
    unsigned long long ga = (unsigned long long)(const void*)(Xc + row0 * DDIM);
    unsigned lds_addr = (unsigned)(size_t)(const void*)Abuf;  // LDS offset (aperture low bits)
    v4u g0 = { 0x1u,                                  // count=1, is_restore=0
               lds_addr,                              // lds_addr
               (unsigned)(ga & 0xFFFFFFFFu),          // global_addr[31:0]
               (unsigned)((ga >> 32) & 0x01FFFFFFu) | 0x80000000u }; // addr[56:32] | type=2
    v8u g1 = { (2u << 16) | (1u << 19),               // data_size=4B, iterate_enable
               1024u << 16,                           // tensor_dim0[15:0]=1024 (bits63:48)
               1u << 16,                              // tensor_dim0 hi=0, tensor_dim1=1
               1024u << 16,                           // tensor_dim1 hi=0, tile_dim0=1024
               1u,                                    // tile_dim1=1, tile_dim2=0
               1024u,                                 // tensor_dim0_stride=1024 dwords
               0u, 0u };                              // dim1 stride unused (tile_dim1=1)
    v4u g2 = { 1u,                                    // tensor_dim2 (unused)
               1028u,                                 // lds_addr_increment  (AST row, dwords)
               1024u,                                 // global_addr_increment (row, dwords)
               15u << 16 };                           // iterate_count = 15 -> 16 iterations
    v4u g3 = { 0u, 0u, 0u, 0u };
    asm volatile("tensor_load_to_lds %0, %1, %2, %3"
                 :: "s"(g0), "s"(g1), "s"(g2), "s"(g3) : "memory");
#if defined(__has_builtin) && __has_builtin(__builtin_amdgcn_s_wait_tensorcnt)
    __builtin_amdgcn_s_wait_tensorcnt(0);
#else
    asm volatile("s_wait_tensorcnt 0x0" ::: "memory");
#endif
  }
#elif USE_ASYNC_LDS
  {
    const char* gsrc = (const char*)(Xc + row0 * DDIM);
    for (int i = tid; i < 16 * (DDIM / 8); i += 256) {
      int r = i >> 8, c = i & 255;
      unsigned lds_off = (unsigned)(size_t)(const void*)(Abuf + (size_t)r * AST + (size_t)c * 8);
      unsigned long long gaddr =
          (unsigned long long)(gsrc + ((size_t)r * DDIM + (size_t)c * 8) * 2);
      asm volatile("global_load_async_to_lds_b128 %0, %1, off"
                   :: "v"(lds_off), "v"(gaddr) : "memory");
    }
    asm volatile("s_wait_asynccnt 0x0" ::: "memory");
  }
#else
  {
    const v4u* src = (const v4u*)(Xc + row0 * DDIM);   // 16*256 v4u
    for (int i = tid; i < 16 * (DDIM / 8); i += 256) {
      int r = i >> 8, c = i & 255;
      *(v4u*)(Abuf + (size_t)r * AST + (size_t)c * 8) = src[(size_t)r * (DDIM / 8) + c];
    }
  }
#endif
  __syncthreads();

  // ---- GEMM1: H[16,HDIM] = A[16,DDIM] @ W1 ; 4 N-tiles per wave per group ----
  for (int g = 0; g < (HDIM / 16) / 32; ++g) {          // 8 groups
    const int nt0 = (g * 8 + wave) * 4;                 // 4 consecutive N-tiles
    v8f acc0 = {0.f,0.f,0.f,0.f,0.f,0.f,0.f,0.f};
    v8f acc1 = acc0, acc2 = acc0, acc3 = acc0;
    const __bf16* b0 = W1T + (size_t)(nt0 * 16 + lm) * DDIM + hh * 16;
    const __bf16* abase = Abuf + (size_t)lm * AST + hh * 16;
#pragma unroll 2
    for (int kt = 0; kt < DDIM; kt += 32) {
      AV a;
      const __bf16* ap = abase + kt;
      a.q[0] = *(const v4u*)ap;          // K: hh*16 .. hh*16+7
      a.q[1] = *(const v4u*)(ap + 8);    // K: hh*16+8 .. hh*16+15
      __builtin_prefetch(b0 + kt + 256, 0, 1);
      AV m0, m1, m2, m3;
      const __bf16* bp = b0 + kt;
      m0.q[0] = *(const v4u*)bp;                        m0.q[1] = *(const v4u*)(bp + 8);
      m1.q[0] = *(const v4u*)(bp + 16 * DDIM);          m1.q[1] = *(const v4u*)(bp + 16 * DDIM + 8);
      m2.q[0] = *(const v4u*)(bp + 32 * DDIM);          m2.q[1] = *(const v4u*)(bp + 32 * DDIM + 8);
      m3.q[0] = *(const v4u*)(bp + 48 * DDIM);          m3.q[1] = *(const v4u*)(bp + 48 * DDIM + 8);
      acc0 = __builtin_amdgcn_wmma_f32_16x16x32_bf16(false, a.v, false, m0.v, (short)0, acc0, false, false);
      acc1 = __builtin_amdgcn_wmma_f32_16x16x32_bf16(false, a.v, false, m1.v, (short)0, acc1, false, false);
      acc2 = __builtin_amdgcn_wmma_f32_16x16x32_bf16(false, a.v, false, m2.v, (short)0, acc2, false, false);
      acc3 = __builtin_amdgcn_wmma_f32_16x16x32_bf16(false, a.v, false, m3.v, (short)0, acc3, false, false);
    }
#pragma unroll
    for (int r = 0; r < 8; ++r) {        // C layout: M = r + hh*8, N = lm
      const size_t rowoff = (size_t)(r + hh * 8) * HST + lm;
      Hbuf[rowoff + (nt0 + 0) * 16] = f2bf(acc0[r]);
      Hbuf[rowoff + (nt0 + 1) * 16] = f2bf(acc1[r]);
      Hbuf[rowoff + (nt0 + 2) * 16] = f2bf(acc2[r]);
      Hbuf[rowoff + (nt0 + 3) * 16] = f2bf(acc3[r]);
    }
  }
  __syncthreads();

  // ---- row norms (f32 partials; b128 LDS reads, 8 bf16 per op) ----
  {
    int m = tid >> 4, g = tid & 15;
    float s = 0.f;
    for (int j = g; j < HDIM / 8; j += 16) {
      B8 t; t.q = *(const v4u*)(Hbuf + (size_t)m * HST + (size_t)j * 8);
#pragma unroll
      for (int e = 0; e < 8; ++e) { float v = bf2f(t.e[e]); s += v * v; }
    }
    red[m * 16 + g] = s;
  }
  __syncthreads();
  if (tid < 16) {
    float s = 0.f;
    for (int g = 0; g < 16; ++g) s += red[tid * 16 + g];
    rnrm[tid] = 1.f / (sqrtf(s) + 1e-5f);
  }
  __syncthreads();
  {
    int m = tid >> 4, g = tid & 15;
    float rn = rnrm[m];
    for (int j = g; j < HDIM / 8; j += 16) {
      __bf16* p = Hbuf + (size_t)m * HST + (size_t)j * 8;
      B8 t; t.q = *(const v4u*)p;
#pragma unroll
      for (int e = 0; e < 8; ++e) t.e[e] = f2bf(bf2f(t.e[e]) * rn);
      *(v4u*)p = t.q;
    }
  }
  __syncthreads();

  // ---- GEMM2: out[16,DDIM] = Hn[16,HDIM] @ W2 ; 4 N-tiles per wave ----
  for (int g = 0; g < (DDIM / 16) / 32; ++g) {          // 4 groups
    const int nt0 = (g * 8 + wave) * 4;
    v8f acc0 = {0.f,0.f,0.f,0.f,0.f,0.f,0.f,0.f};
    v8f acc1 = acc0, acc2 = acc0, acc3 = acc0;
    const __bf16* b0 = W2T + (size_t)(nt0 * 16 + lm) * HDIM + hh * 16;
    const __bf16* abase = Hbuf + (size_t)lm * HST + hh * 16;
#pragma unroll 2
    for (int kt = 0; kt < HDIM; kt += 32) {
      AV a;
      const __bf16* ap = abase + kt;
      a.q[0] = *(const v4u*)ap;
      a.q[1] = *(const v4u*)(ap + 8);
      __builtin_prefetch(b0 + kt + 256, 0, 1);
      AV m0, m1, m2, m3;
      const __bf16* bp = b0 + kt;
      m0.q[0] = *(const v4u*)bp;                        m0.q[1] = *(const v4u*)(bp + 8);
      m1.q[0] = *(const v4u*)(bp + 16 * HDIM);          m1.q[1] = *(const v4u*)(bp + 16 * HDIM + 8);
      m2.q[0] = *(const v4u*)(bp + 32 * HDIM);          m2.q[1] = *(const v4u*)(bp + 32 * HDIM + 8);
      m3.q[0] = *(const v4u*)(bp + 48 * HDIM);          m3.q[1] = *(const v4u*)(bp + 48 * HDIM + 8);
      acc0 = __builtin_amdgcn_wmma_f32_16x16x32_bf16(false, a.v, false, m0.v, (short)0, acc0, false, false);
      acc1 = __builtin_amdgcn_wmma_f32_16x16x32_bf16(false, a.v, false, m1.v, (short)0, acc1, false, false);
      acc2 = __builtin_amdgcn_wmma_f32_16x16x32_bf16(false, a.v, false, m2.v, (short)0, acc2, false, false);
      acc3 = __builtin_amdgcn_wmma_f32_16x16x32_bf16(false, a.v, false, m3.v, (short)0, acc3, false, false);
    }
    float* obase = out + row0 * DDIM + (size_t)nt0 * 16 + lm;
#pragma unroll
    for (int r = 0; r < 8; ++r) {
      float* orow = obase + (size_t)(r + hh * 8) * DDIM;
      orow[0]  = acc0[r];
      orow[16] = acc1[r];
      orow[32] = acc2[r];
      orow[48] = acc3[r];
    }
  }
}

// ---------------------------------------------------------------------------
// Host launcher
// ---------------------------------------------------------------------------
extern "C" void kernel_launch(void* const* d_in, const int* in_sizes, int n_in,
                              void* d_out, int out_size, void* d_ws, size_t ws_size,
                              hipStream_t stream) {
  const float* X  = (const float*)d_in[0];   // [B,S,D]
  const float* W1 = (const float*)d_in[1];   // [D,H]
  const float* W2 = (const float*)d_in[2];   // [H,D]
  float* out = (float*)d_out;                // [B,S,D]

  // Workspace layout (bytes):
  //   sums : BATCH*NCHUNK*DDIM f32                =  2 MiB
  //   Xc   : BATCH*SEQ*DDIM bf16                  = 64 MiB
  //   W1T  : HDIM*DDIM bf16                       = 16 MiB
  //   W2T  : DDIM*HDIM bf16                       = 16 MiB
  char* ws = (char*)d_ws;
  float*  sums = (float*)ws;
  __bf16* Xc   = (__bf16*)(ws + (size_t)2 * 1024 * 1024);
  __bf16* W1T  = (__bf16*)(ws + (size_t)66 * 1024 * 1024);
  __bf16* W2T  = (__bf16*)(ws + (size_t)82 * 1024 * 1024);

  dim3 tb(32, 8);
  // W1 [D,H] -> W1T [H,D]
  transpose_f32_to_bf16<<<dim3(HDIM / 32, DDIM / 32), tb, 0, stream>>>(W1, W1T, DDIM, HDIM);
  // W2 [H,D] -> W2T [D,H]
  transpose_f32_to_bf16<<<dim3(DDIM / 32, HDIM / 32), tb, 0, stream>>>(W2, W2T, HDIM, DDIM);

  chunk_sums<<<BATCH * NCHUNK, 256, 0, stream>>>(X, sums);
  scan_apply<<<BATCH * NCHUNK, 256, 0, stream>>>(X, sums, Xc);

  const size_t smem = (size_t)16 * AST * 2 + (size_t)16 * HST * 2 + 256 * 4 + 16 * 4;
  fused_mlp<<<(BATCH * SEQ) / 16, 256, smem, stream>>>(Xc, W1T, W2T, out);
}
